// GatedAttnPool_21294447853999
// MI455X (gfx1250) — matrix operand order repaired and checked by reading, compiled, not run on hardware
//
#include <hip/hip_runtime.h>
#include <hip/hip_bf16.h>

// GatedAttnPool for MI455X (gfx1250, wave32).
// Gate GEMM (131072x512 @ 512x512) uses f16 split precision (hi/lo) with
// V_WMMA_F32_16X16X32_F16: 3 products (hh, lh, hl) give ~f32 accuracy at
// 16-bit WMMA throughput. B tiles (pre-split f16 W1T) are staged into LDS
// by the Tensor Data Mover (TENSOR_LOAD_TO_LDS + s_wait_tensorcnt), using
// the D# pad fields to produce the bank-conflict-avoiding 528-half stride.
// Pool pass streams x once more (memory-bound, ~11us at 23.3 TB/s).

#define DIM   512
#define BATCH 4
#define RDIM  32
#define TDIM  1024
#define NROWS (BATCH * RDIM * TDIM)   // 131072 rows of x
#define EPS_LN 1e-5f
#define HSTR  528   // LDS row stride in halves; 1056B = 32 mod 256 -> spread banks

typedef __attribute__((ext_vector_type(16))) _Float16 v16h;
typedef __attribute__((ext_vector_type(8)))  _Float16 v8h;
typedef __attribute__((ext_vector_type(4)))  _Float16 v4h;
typedef __attribute__((ext_vector_type(4)))  float    v4f;
typedef __attribute__((ext_vector_type(8)))  float    v8f;
typedef __attribute__((ext_vector_type(4)))  unsigned int v4u;
typedef __attribute__((ext_vector_type(8)))  int      v8i;
typedef __attribute__((ext_vector_type(4)))  int      v4i;

// LDS byte offsets of the B tiles inside the dynamic allocation
#define BHI_BYTE_OFF (2u * 128u * HSTR * 2u)            // after Ahi+Alo = 270336
#define BLO_BYTE_OFF (BHI_BYTE_OFF + 16u * HSTR * 2u)   // 287232

#if defined(__has_builtin)
#  if __has_builtin(__builtin_amdgcn_tensor_load_to_lds) && \
      __has_builtin(__builtin_amdgcn_s_wait_tensorcnt)
#    define USE_TDM 1
#  endif
#endif
#ifndef USE_TDM
#  define USE_TDM 0
#endif

#if USE_TDM
// TDM: DMA `rows`(=4) x 512 halves from gsrc (row stride 512 halves) into LDS
// at lds_byte_off, inserting 32B padding after each 1024B row (-> 528-half rows).
__device__ __forceinline__ void tdm_load_rows(const _Float16* gsrc,
                                              unsigned lds_byte_off) {
    const unsigned long long ga = (unsigned long long)gsrc;
    // D# group 0: count=1 | lds_addr | global_addr[56:0] | type=2
    v4u g0 = { 1u,
               lds_byte_off,
               (unsigned)(ga & 0xffffffffull),
               (unsigned)((ga >> 32) & 0x1ffffffull) | (2u << 30) };
    // D# group 1: data_size=2B(code1), pad_enable, pad_interval=7(256 DW),
    // pad_amount=7(8 DW=32B); tensor_dim0=512, tensor_dim1=4,
    // tile_dim0=512, tile_dim1=4, tensor_dim0_stride=512.
    v8i g1 = { (int)((1u << 16) | (1u << 20) | (7u << 22) | (7u << 25)),
               (int)(512u << 16),    // tensor_dim0[15:0] @ bits 63:48
               (int)(4u   << 16),    // tensor_dim1[15:0] @ bits 95:80
               (int)(512u << 16),    // tile_dim0 @ bits 127:112
               4,                    // tile_dim1 @ bits 143:128
               512,                  // tensor_dim0_stride[31:0] @ bits 191:160
               0, 0 };
    v4i z4 = { 0, 0, 0, 0 };
#if __clang_major__ >= 23
    v8i z8 = { 0, 0, 0, 0, 0, 0, 0, 0 };
    __builtin_amdgcn_tensor_load_to_lds(g0, g1, z4, z4, z8, 0);
#else
    __builtin_amdgcn_tensor_load_to_lds(g0, g1, z4, z4, 0);
#endif
}
#endif

// Load A operand: lane's two 16-byte K-chunks -> v16h
// (contiguous K order used here for A and B: lanes0-15 K=[0..15], lanes16-31 K=[16..31])
__device__ __forceinline__ v16h load_a16(const _Float16* p) {
    v8h c0 = *(const v8h*)p;
    v8h c1 = *(const v8h*)(p + 8);
    return __builtin_shufflevector(c0, c1, 0, 1, 2, 3, 4, 5, 6, 7,
                                   8, 9, 10, 11, 12, 13, 14, 15);
}

// ---------------------------------------------------------------------------
// K0a: transpose bottom half of W1 and split into f16 hi/lo:
//   v = W1[(512+k)*512 + n];  W1Th[n][k] = f16(v); W1Tl[n][k] = f16(v - hi)
// ---------------------------------------------------------------------------
__global__ void transpose_w1_kernel(const float* __restrict__ W1,
                                    _Float16* __restrict__ W1Th,
                                    _Float16* __restrict__ W1Tl) {
    __shared__ float tile[32][33];
    const int n0 = blockIdx.x * 32;
    const int k0 = blockIdx.y * 32;
    const int tx = threadIdx.x;       // 0..31
    const int ty = threadIdx.y;       // 0..7
    for (int i = ty; i < 32; i += 8)
        tile[i][tx] = W1[(size_t)(DIM + k0 + i) * DIM + n0 + tx];
    __syncthreads();
    for (int i = ty; i < 32; i += 8) {
        const float v = tile[tx][i];
        const _Float16 h = (_Float16)v;
        const _Float16 l = (_Float16)(v - (float)h);
        W1Th[(size_t)(n0 + i) * DIM + k0 + tx] = h;
        W1Tl[(size_t)(n0 + i) * DIM + k0 + tx] = l;
    }
}

// ---------------------------------------------------------------------------
// K0b: tb[b][n] = b1[n] + sum_k text_query[b][k] * W1[k][n]   (tiny, f32)
// ---------------------------------------------------------------------------
__global__ void text_bias_kernel(const float* __restrict__ tq,
                                 const float* __restrict__ W1,
                                 const float* __restrict__ b1,
                                 float* __restrict__ tb) {
    const int b = blockIdx.x;
    const int n = threadIdx.x;        // 512 threads
    float s = b1[n];
    for (int k = 0; k < DIM; ++k)
        s += tq[b * DIM + k] * W1[(size_t)k * DIM + n];
    tb[b * DIM + n] = s;
}

// ---------------------------------------------------------------------------
// K1: gate[row] = b2 + sum_n relu( (x_row . W1T[n]) + tb[b][n] ) * W2[n]
// Block = 128 threads (4 waves); wave owns 32 rows = two 16-row WMMA A-tiles.
// x converted to f16 hi/lo during LDS staging (no extra HBM traffic).
// LDS (halves): Ahi[128][528] Alo[128][528] Bhi[16][528] Blo[16][528] = 297KB
// ---------------------------------------------------------------------------
__global__ void gate_scores_kernel(const float* __restrict__ x,
                                   const _Float16* __restrict__ W1Th,
                                   const _Float16* __restrict__ W1Tl,
                                   const float* __restrict__ tb,
                                   const float* __restrict__ W2,
                                   const float* __restrict__ b2,
                                   float* __restrict__ gate) {
    extern __shared__ _Float16 smem[];
    _Float16* Ahi = smem;                    // 128*HSTR
    _Float16* Alo = Ahi + 128 * HSTR;
    _Float16* Bhi = Alo + 128 * HSTR;        // 16*HSTR
    _Float16* Blo = Bhi + 16 * HSTR;

    const int tid  = threadIdx.x;
    const int lane = tid & 31;
    const int wave = tid >> 5;               // 0..3
    const int rowBase = blockIdx.x * 128;
    const int bIdx = rowBase / (RDIM * TDIM);   // batch constant per block

    // ---- Stage A: 32 rows per wave; load f32, split to f16 hi/lo, store LDS
    for (int m = 0; m < 32; ++m) {
        const int rloc = wave * 32 + m;
        const float* src = x + (size_t)(rowBase + rloc) * DIM;
        _Float16* dh = Ahi + rloc * HSTR;
        _Float16* dl = Alo + rloc * HSTR;
        for (int j = lane; j < DIM / 4; j += 32) {
            const v4f v = *(const v4f*)(src + j * 4);
            v4h hv, lv;
#pragma unroll
            for (int e = 0; e < 4; ++e) {
                const _Float16 h = (_Float16)v[e];
                hv[e] = h;
                lv[e] = (_Float16)(v[e] - (float)h);
            }
            *(v4h*)(dh + j * 4) = hv;
            *(v4h*)(dl + j * 4) = lv;
        }
    }
    __syncthreads();

    const int half = lane >> 4;              // 0 | 1
    const int l16  = lane & 15;
    const int aOff = 16 * half;              // lane's first A K-chunk offset
    const _Float16* a0h = Ahi + (wave * 32 + l16) * HSTR + aOff;
    const _Float16* a0l = Alo + (wave * 32 + l16) * HSTR + aOff;
    const _Float16* a1h = Ahi + (wave * 32 + 16 + l16) * HSTR + aOff;
    const _Float16* a1l = Alo + (wave * 32 + 16 + l16) * HSTR + aOff;
    // B layout: lane = column n=l16; contiguous K run of 16 halves
    const _Float16* bhp = Bhi + l16 * HSTR + 16 * half;
    const _Float16* blp = Blo + l16 * HSTR + 16 * half;

#if USE_TDM
    // Scalar strip index for the per-wave TDM descriptor (must be in SGPRs)
    const int wq = __builtin_amdgcn_readfirstlane(tid >> 5);
#endif

    v8f acc0 = {}; v8f acc1 = {};

    for (int nt = 0; nt < DIM / 16; ++nt) {
        __syncthreads();                     // Bs consumed by previous iteration
#if USE_TDM
        // ---- Stage B via Tensor Data Mover: each wave DMAs its 4-row strip
        // of W1Th/W1Tl; D# pad fields insert 32B per 1024B row -> HSTR=528.
        tdm_load_rows(W1Th + (size_t)(nt * 16 + wq * 4) * DIM,
                      BHI_BYTE_OFF + (unsigned)(wq * 4 * HSTR * 2));
        tdm_load_rows(W1Tl + (size_t)(nt * 16 + wq * 4) * DIM,
                      BLO_BYTE_OFF + (unsigned)(wq * 4 * HSTR * 2));
        __builtin_amdgcn_s_wait_tensorcnt(0);
#else
        // ---- Stage B fallback: cooperative 16B-chunk copy
        for (int idx = tid; idx < 2 * 16 * (DIM / 8); idx += 128) {
            const int comp = idx >> 10;          // 0=hi, 1=lo (16*64 = 1024)
            const int rrow = (idx & 1023) >> 6;  // 0..15
            const int cc   = idx & 63;           // 0..63 chunks of 8 halves
            const _Float16* gsrc =
                (comp ? W1Tl : W1Th) + (size_t)(nt * 16 + rrow) * DIM + cc * 8;
            _Float16* ldst = (comp ? Blo : Bhi) + rrow * HSTR + cc * 8;
            *(v8h*)ldst = *(const v8h*)gsrc;
        }
#endif
        __syncthreads();

        v8f c0 = {}; v8f c1 = {};
        for (int k = 0; k < DIM; k += 32) {
            const v16h A0h = load_a16(a0h + k);
            const v16h A0l = load_a16(a0l + k);
            const v16h A1h = load_a16(a1h + k);
            const v16h A1l = load_a16(a1l + k);
            const v16h Bh  = *(const v16h*)(bhp + k);
            const v16h Bl  = *(const v16h*)(blp + k);
            // x.w ~= xh.wh + xl.wh + xh.wl  (f32 accumulate)
            c0 = __builtin_amdgcn_wmma_f32_16x16x32_f16(
                    false, A0h, false, Bh, (short)0, c0, false, false);
            c1 = __builtin_amdgcn_wmma_f32_16x16x32_f16(
                    false, A1h, false, Bh, (short)0, c1, false, false);
            c0 = __builtin_amdgcn_wmma_f32_16x16x32_f16(
                    false, A0l, false, Bh, (short)0, c0, false, false);
            c1 = __builtin_amdgcn_wmma_f32_16x16x32_f16(
                    false, A1l, false, Bh, (short)0, c1, false, false);
            c0 = __builtin_amdgcn_wmma_f32_16x16x32_f16(
                    false, A0h, false, Bl, (short)0, c0, false, false);
            c1 = __builtin_amdgcn_wmma_f32_16x16x32_f16(
                    false, A1h, false, Bl, (short)0, c1, false, false);
        }

        // Fused epilogue: +text/bias, relu, dot with W2 (per-lane column scalar)
        const float tbv = tb[bIdx * DIM + nt * 16 + l16];
        const float w2v = W2[nt * 16 + l16];
#pragma unroll
        for (int i = 0; i < 8; ++i) {
            const float h0 = c0[i] + tbv;
            const float h1 = c1[i] + tbv;
            acc0[i] += (h0 > 0.f ? h0 : 0.f) * w2v;
            acc1[i] += (h1 > 0.f ? h1 : 0.f) * w2v;
        }
    }

    // ---- Reduce over the 16 lanes of each half-wave; write gate scores.
    // C layout: VGPR i -> row i (lanes0-15), row i+8 (lanes16-31).
    const float bias2 = b2[0];
    const int tileRow = rowBase + wave * 32;
#pragma unroll
    for (int i = 0; i < 8; ++i) {
        float s0 = acc0[i], s1 = acc1[i];
        for (int m = 1; m <= 8; m <<= 1) {
            s0 += __shfl_xor(s0, m, 32);
            s1 += __shfl_xor(s1, m, 32);
        }
        if (lane == 0) {
            gate[tileRow + i]      = s0 + bias2;   // tile0 rows 0..7
            gate[tileRow + 16 + i] = s1 + bias2;   // tile1 rows 0..7
        }
        if (lane == 16) {
            gate[tileRow + 8 + i]  = s0 + bias2;   // tile0 rows 8..15
            gate[tileRow + 24 + i] = s1 + bias2;   // tile1 rows 8..15
        }
    }
}

// ---------------------------------------------------------------------------
// K2: per (b,t): masked softmax over r, weighted pool of x, LayerNorm.
// Block = 512 threads (one per d), grid = b*t = 4096.
// ---------------------------------------------------------------------------
__global__ void pool_ln_kernel(const float* __restrict__ x,
                               const float* __restrict__ gate,
                               const unsigned char* __restrict__ mask, // jax bool
                               const float* __restrict__ gamma,
                               const float* __restrict__ beta,
                               float* __restrict__ out) {
    __shared__ float wts[RDIM];
    __shared__ float redS[16];
    __shared__ float redQ[16];
    __shared__ float sMean, sRstd;

    const int bt = blockIdx.x;
    const int b = bt >> 10;
    const int t = bt & (TDIM - 1);
    const int tid = threadIdx.x;     // 0..511

    if (tid < RDIM) {                // wave0: softmax over r in registers
        float g = gate[((size_t)(b * RDIM + tid)) * TDIM + t];
        if (mask[b * RDIM + tid] == 0) g = -10000.0f;
        float mx = g;
        for (int s = 16; s >= 1; s >>= 1) mx = fmaxf(mx, __shfl_xor(mx, s, 32));
        float e = __expf(g - mx);
        float sum = e;
        for (int s = 16; s >= 1; s >>= 1) sum += __shfl_xor(sum, s, 32);
        wts[tid] = e / sum;
    }
    __syncthreads();

    // Weighted pool over r (coalesced: 512 threads read 2048B runs per r)
    float acc = 0.f;
    for (int r = 0; r < RDIM; ++r)
        acc += wts[r] * x[(((size_t)(b * RDIM + r)) * TDIM + t) * DIM + tid];

    // LayerNorm: block reduction of sum & sumsq
    float s = acc, q = acc * acc;
    for (int m = 16; m >= 1; m >>= 1) {
        s += __shfl_xor(s, m, 32);
        q += __shfl_xor(q, m, 32);
    }
    if ((tid & 31) == 0) { redS[tid >> 5] = s; redQ[tid >> 5] = q; }
    __syncthreads();
    if (tid < 16) {
        float ss = redS[tid], qq = redQ[tid];
        for (int m = 8; m >= 1; m >>= 1) {
            ss += __shfl_xor(ss, m, 32);
            qq += __shfl_xor(qq, m, 32);
        }
        if (tid == 0) {
            const float mean = ss * (1.0f / DIM);
            const float var  = qq * (1.0f / DIM) - mean * mean;
            sMean = mean;
            sRstd = rsqrtf(var + EPS_LN);
        }
    }
    __syncthreads();

    out[(size_t)bt * DIM + tid] =
        (acc - sMean) * sRstd * gamma[tid] + beta[tid];
}

// ---------------------------------------------------------------------------
extern "C" void kernel_launch(void* const* d_in, const int* in_sizes, int n_in,
                              void* d_out, int out_size, void* d_ws, size_t ws_size,
                              hipStream_t stream) {
    const float* x     = (const float*)d_in[0];   // [4,32,1024,512]
    const float* tq    = (const float*)d_in[1];   // [4,512]
    const unsigned char* mask = (const unsigned char*)d_in[2]; // [4,32] bool
    const float* W1    = (const float*)d_in[3];   // [1024,512]
    const float* b1    = (const float*)d_in[4];   // [512]
    const float* W2    = (const float*)d_in[5];   // [512,1]
    const float* b2    = (const float*)d_in[6];   // [1]
    const float* gamma = (const float*)d_in[7];   // [512]
    const float* beta  = (const float*)d_in[8];   // [512]
    float* out = (float*)d_out;                   // [4,1024,512]

    // Workspace: tb f32[2048] | gate f32[131072] | W1Th f16[262144] | W1Tl f16
    float* tb   = (float*)d_ws;
    float* gate = tb + BATCH * DIM;
    _Float16* W1Th = (_Float16*)(gate + NROWS);
    _Float16* W1Tl = W1Th + DIM * DIM;

    transpose_w1_kernel<<<dim3(16, 16), dim3(32, 8), 0, stream>>>(W1, W1Th, W1Tl);
    text_bias_kernel<<<BATCH, DIM, 0, stream>>>(tq, W1, b1, tb);

    const size_t ldsBytes =
        (size_t)(2 * 128 * HSTR + 2 * 16 * HSTR) * sizeof(_Float16); // 304,128 B
    gate_scores_kernel<<<NROWS / 128, 128, ldsBytes, stream>>>(
        x, W1Th, W1Tl, tb, W2, b2, gate);

    pool_ln_kernel<<<BATCH * TDIM, DIM, 0, stream>>>(
        x, gate, mask, gamma, beta, out);
}
